// SkillPrior_7851200217363
// MI455X (gfx1250) — compile-verified, matrix operand order checked by reading, not run on hardware
//
// SkillPrior (LSTM VAE) for MI455X / gfx1250 — bf16 WMMA recurrence, fp32 epilogue.
#include <hip/hip_runtime.h>

typedef __attribute__((ext_vector_type(16))) __bf16 v16bf;
typedef __attribute__((ext_vector_type(8)))  __bf16 v8bf;
typedef __attribute__((ext_vector_type(8)))  float  v8f;

namespace {

constexpr int B_   = 2048;
constexpr int T_   = 50;
constexpr int Zd   = 128;
constexpr int HIDc = 1024;
constexpr int SDIMc= 60;
constexpr int G4   = 4096;   // 4*HID

__device__ __forceinline__ unsigned short f2bf(float f) {
  unsigned int u = __float_as_uint(f);
  u += 0x7fffu + ((u >> 16) & 1u);          // round-to-nearest-even
  return (unsigned short)(u >> 16);
}

__device__ __forceinline__ float sigmoidf_(float x) {
  return 1.0f / (1.0f + __expf(-x));
}

__device__ __forceinline__ float softplusf_(float x) {
  return fmaxf(x, 0.0f) + log1pf(__expf(-fabsf(x)));
}

// Load one 16-lane-parallel x K=32 bf16 WMMA fragment.
// Per ISA layout: lanes 0-15 hold K={0..7,16..23}, lanes 16-31 hold K={8..15,24..31};
// caller pre-offsets p by (row*ld + (lane&16 ? 8 : 0) + k0) elements.
__device__ __forceinline__ v16bf ldfrag(const __bf16* p) {
  v8bf lo = *(const v8bf*)(p);       // K base + 0..7
  v8bf hi = *(const v8bf*)(p + 16);  // K base + 16..23
  return __builtin_shufflevector(lo, hi, 0,1,2,3,4,5,6,7,8,9,10,11,12,13,14,15);
}

// ---------------- weight conversion fp32 -> bf16 ----------------
__global__ void conv_k(const float* __restrict__ w0, const float* __restrict__ w1,
                       unsigned short* __restrict__ o0, unsigned short* __restrict__ o1) {
  int id = blockIdx.x * blockDim.x + threadIdx.x;
  if (id >= G4 * HIDc) return;
  o0[id] = f2bf(w0[id]);
  o1[id] = f2bf(w1[id]);
}

// ---------------- zero h (bf16), h (f32), c ----------------
__global__ void init_k(unsigned short* __restrict__ hbf, float* __restrict__ hf,
                       float* __restrict__ c) {
  int id = blockIdx.x * blockDim.x + threadIdx.x;
  if (id >= B_ * HIDc) return;
  hbf[id] = 0;
  hf[id]  = 0.0f;
  c[id]   = 0.0f;
}

// ---------------- Zst = states @ Ws.T + bs ; split mean / softplus(var) ----------------
__global__ void zst_k(const float* __restrict__ states, const float* __restrict__ Ws,
                      const float* __restrict__ bs, float* __restrict__ zs_mean,
                      float* __restrict__ zs_var) {
  int id = blockIdx.x * blockDim.x + threadIdx.x;
  if (id >= B_ * 2 * Zd) return;
  int b  = id >> 8;          // /256
  int nn = id & 255;
  const float* sr = states + (size_t)b * SDIMc;
  const float* wr = Ws + (size_t)nn * SDIMc;
  float s = bs[nn];
  for (int k = 0; k < SDIMc; ++k) s += sr[k] * wr[k];
  if (nn < Zd) zs_mean[(size_t)b * Zd + nn] = s;
  else         zs_var[(size_t)b * Zd + (nn - Zd)] = softplusf_(s);
}

// ---------------- fused LSTM step: g = pre + h @ W_hh.T ; gate math ----------------
// grid = (HID/128, B/32), block = 256 (8 waves). Wave tile: 32 rows x 16 cols x 4 gates.
__global__ __launch_bounds__(256) void lstm_step_k(
    const unsigned short* __restrict__ h_in_bf,   // (B, HID) bf16
    const unsigned short* __restrict__ whh_bf,    // (4*HID, HID) bf16
    float* __restrict__ c_io,                     // (B, HID) in-place
    const float* __restrict__ pre_dec,            // (B, 4*HID) decoder mode if non-null
    const float* __restrict__ actions,            // (B, T, 2) encoder mode
    const float* __restrict__ We_ih,              // (4*HID, 2)
    const float* __restrict__ be,                 // (4*HID)
    int t,
    unsigned short* __restrict__ h_out_bf,        // (B, HID) bf16
    float* __restrict__ h_out_f32)                // (B, HID) f32
{
  const int wave  = threadIdx.x >> 5;
  const int lane  = threadIdx.x & 31;
  const int n0    = blockIdx.x * 128 + wave * 16;
  const int m0    = blockIdx.y * 32;
  const int lrow  = lane & 15;
  const int lhalf = (lane & 16) ? 8 : 0;

  const __bf16* ap0 = (const __bf16*)h_in_bf + (size_t)(m0 + lrow) * HIDc + lhalf;
  const __bf16* ap1 = ap0 + (size_t)16 * HIDc;
  const __bf16* bp  = (const __bf16*)whh_bf + (size_t)(n0 + lrow) * HIDc + lhalf;

  v8f acc[4][2] = {};

  for (int k0 = 0; k0 < HIDc; k0 += 32) {
    v16bf a0 = ldfrag(ap0 + k0);
    v16bf a1 = ldfrag(ap1 + k0);
#pragma unroll
    for (int g = 0; g < 4; ++g) {
      v16bf bfr = ldfrag(bp + (size_t)g * HIDc * HIDc + k0);
      acc[g][0] = __builtin_amdgcn_wmma_f32_16x16x32_bf16(
          false, a0, false, bfr, (short)0, acc[g][0], false, false);
      acc[g][1] = __builtin_amdgcn_wmma_f32_16x16x32_bf16(
          false, a1, false, bfr, (short)0, acc[g][1], false, false);
    }
  }

  // Epilogue: C/D layout -> element (m, n); gate math; write h/c.
  const int n = n0 + lrow;
#pragma unroll
  for (int s = 0; s < 2; ++s) {
    const int mbase = m0 + s * 16 + lhalf;   // lanes 16-31 own M=8..15 of each tile
#pragma unroll
    for (int r = 0; r < 8; ++r) {
      const int m = mbase + r;
      float pi, pf, pg, po;
      if (pre_dec) {
        const float* pb = pre_dec + (size_t)m * G4 + n;
        pi = pb[0]; pf = pb[HIDc]; pg = pb[2 * HIDc]; po = pb[3 * HIDc];
      } else {
        const float ax = actions[(size_t)m * (T_ * 2) + t * 2 + 0];
        const float ay = actions[(size_t)m * (T_ * 2) + t * 2 + 1];
        pi = ax * We_ih[(size_t)(0 * HIDc + n) * 2] + ay * We_ih[(size_t)(0 * HIDc + n) * 2 + 1] + be[0 * HIDc + n];
        pf = ax * We_ih[(size_t)(1 * HIDc + n) * 2] + ay * We_ih[(size_t)(1 * HIDc + n) * 2 + 1] + be[1 * HIDc + n];
        pg = ax * We_ih[(size_t)(2 * HIDc + n) * 2] + ay * We_ih[(size_t)(2 * HIDc + n) * 2 + 1] + be[2 * HIDc + n];
        po = ax * We_ih[(size_t)(3 * HIDc + n) * 2] + ay * We_ih[(size_t)(3 * HIDc + n) * 2 + 1] + be[3 * HIDc + n];
      }
      const float gI = sigmoidf_(acc[0][s][r] + pi);
      const float gF = sigmoidf_(acc[1][s][r] + pf);
      const float gG = tanhf   (acc[2][s][r] + pg);
      const float gO = sigmoidf_(acc[3][s][r] + po);
      const size_t idx = (size_t)m * HIDc + n;
      const float cN = gF * c_io[idx] + gI * gG;
      const float hN = gO * tanhf(cN);
      c_io[idx]      = cN;
      h_out_f32[idx] = hN;
      h_out_bf[idx]  = f2bf(hN);
    }
  }
}

// ---------------- Zseq = hT @ Wpe.T + bpe ----------------
__global__ void zseq_k(const float* __restrict__ h, const float* __restrict__ Wpe,
                       const float* __restrict__ bpe, float* __restrict__ zseq) {
  int id = blockIdx.x * blockDim.x + threadIdx.x;
  if (id >= B_ * 2 * Zd) return;
  int b  = id >> 8;
  int nn = id & 255;
  const float* hr = h + (size_t)b * HIDc;
  const float* wr = Wpe + (size_t)nn * HIDc;
  float s = bpe[nn];
  for (int k = 0; k < HIDc; ++k) s += hr[k] * wr[k];
  zseq[id] = s;
}

// ---------------- z_mean, z_var=softplus, z = mean + var*eps ----------------
__global__ void z_k(const float* __restrict__ zseq, const float* __restrict__ eps,
                    float* __restrict__ z_mean, float* __restrict__ z_var,
                    float* __restrict__ z_out, float* __restrict__ z_ws) {
  int id = blockIdx.x * blockDim.x + threadIdx.x;
  if (id >= B_ * Zd) return;
  int b = id >> 7;
  int j = id & 127;
  float m = zseq[(size_t)b * 2 * Zd + j];
  float v = softplusf_(zseq[(size_t)b * 2 * Zd + Zd + j]);
  float zz = m + v * eps[id];
  z_mean[id] = m;
  z_var[id]  = v;
  z_out[id]  = zz;
  z_ws[id]   = zz;
}

// ---------------- pre_d = z @ Wd_ih.T + bd ----------------
__global__ void pred_k(const float* __restrict__ z, const float* __restrict__ Wd_ih,
                       const float* __restrict__ bd, float* __restrict__ pre_d) {
  int id = blockIdx.x * blockDim.x + threadIdx.x;
  if (id >= B_ * G4) return;
  int b  = id >> 12;       // /4096
  int nn = id & 4095;
  const float* zr = z + (size_t)b * Zd;
  const float* wr = Wd_ih + (size_t)nn * Zd;
  float s = bd[nn];
  for (int k = 0; k < Zd; ++k) s += zr[k] * wr[k];
  pre_d[id] = s;
}

// ---------------- y[t] = h @ Wpd.T + bpd (N=2) ; one wave per row ----------------
__global__ __launch_bounds__(256) void yproj_k(const float* __restrict__ h,
                                               const float* __restrict__ Wpd,
                                               const float* __restrict__ bpd,
                                               float* __restrict__ X, int t) {
  const int wave = threadIdx.x >> 5;
  const int lane = threadIdx.x & 31;
  const int b    = blockIdx.x * 8 + wave;
  const float* hr = h + (size_t)b * HIDc;
  float s0 = 0.0f, s1 = 0.0f;
  for (int k = lane; k < HIDc; k += 32) {
    float hv = hr[k];
    s0 += hv * Wpd[k];
    s1 += hv * Wpd[HIDc + k];
  }
  for (int off = 16; off > 0; off >>= 1) {
    s0 += __shfl_xor(s0, off, 32);
    s1 += __shfl_xor(s1, off, 32);
  }
  if (lane == 0) {
    X[(size_t)b * (T_ * 2) + t * 2 + 0] = s0 + bpd[0];
    X[(size_t)b * (T_ * 2) + t * 2 + 1] = s1 + bpd[1];
  }
}

} // anonymous namespace

extern "C" void kernel_launch(void* const* d_in, const int* in_sizes, int n_in,
                              void* d_out, int out_size, void* d_ws, size_t ws_size,
                              hipStream_t stream) {
  (void)in_sizes; (void)n_in; (void)out_size; (void)ws_size;

  const float* states = (const float*)d_in[0];
  const float* actions= (const float*)d_in[1];
  const float* eps    = (const float*)d_in[2];
  const float* Ws     = (const float*)d_in[3];
  const float* bs     = (const float*)d_in[4];
  const float* We_ih  = (const float*)d_in[5];
  const float* We_hh  = (const float*)d_in[6];
  const float* be     = (const float*)d_in[7];
  const float* Wpe    = (const float*)d_in[8];
  const float* bpe    = (const float*)d_in[9];
  const float* Wd_ih  = (const float*)d_in[10];
  const float* Wd_hh  = (const float*)d_in[11];
  const float* bd     = (const float*)d_in[12];
  const float* Wpd    = (const float*)d_in[13];
  const float* bpd    = (const float*)d_in[14];
  float* out = (float*)d_out;

  // workspace carve-up
  char* wsb = (char*)d_ws;
  size_t off = 0;
  auto take = [&](size_t bytes) -> void* {
    void* p = wsb + off;
    off = (off + bytes + 255) & ~(size_t)255;
    return p;
  };
  unsigned short* wehh_bf = (unsigned short*)take((size_t)G4 * HIDc * 2);
  unsigned short* wdhh_bf = (unsigned short*)take((size_t)G4 * HIDc * 2);
  unsigned short* hbf0    = (unsigned short*)take((size_t)B_ * HIDc * 2);
  unsigned short* hbf1    = (unsigned short*)take((size_t)B_ * HIDc * 2);
  float* hf32 = (float*)take((size_t)B_ * HIDc * 4);
  float* cbuf = (float*)take((size_t)B_ * HIDc * 4);
  float* zseq = (float*)take((size_t)B_ * 2 * Zd * 4);
  float* zbuf = (float*)take((size_t)B_ * Zd * 4);
  float* pred = (float*)take((size_t)B_ * G4 * 4);

  const size_t NZ = (size_t)B_ * Zd;
  float* o_zmean  = out;
  float* o_zvar   = out + NZ;
  float* o_zsmean = out + 2 * NZ;
  float* o_zsvar  = out + 3 * NZ;
  float* o_z      = out + 4 * NZ;
  float* o_X      = out + 5 * NZ;

  const int TPB = 256;
  { int n = G4 * HIDc;   conv_k<<<(n + TPB - 1) / TPB, TPB, 0, stream>>>(We_hh, Wd_hh, wehh_bf, wdhh_bf); }
  { int n = B_ * 2 * Zd; zst_k <<<(n + TPB - 1) / TPB, TPB, 0, stream>>>(states, Ws, bs, o_zsmean, o_zsvar); }
  { int n = B_ * HIDc;   init_k<<<(n + TPB - 1) / TPB, TPB, 0, stream>>>(hbf0, hf32, cbuf); }

  dim3 lgrid(HIDc / 128, B_ / 32);
  for (int t = 0; t < T_; ++t) {
    unsigned short* hin  = (t & 1) ? hbf1 : hbf0;
    unsigned short* hout = (t & 1) ? hbf0 : hbf1;
    lstm_step_k<<<lgrid, TPB, 0, stream>>>(hin, wehh_bf, cbuf, nullptr,
                                           actions, We_ih, be, t, hout, hf32);
  }

  { int n = B_ * 2 * Zd; zseq_k<<<(n + TPB - 1) / TPB, TPB, 0, stream>>>(hf32, Wpe, bpe, zseq); }
  { int n = B_ * Zd;     z_k   <<<(n + TPB - 1) / TPB, TPB, 0, stream>>>(zseq, eps, o_zmean, o_zvar, o_z, zbuf); }
  { int n = B_ * G4;     pred_k<<<(n + TPB - 1) / TPB, TPB, 0, stream>>>(zbuf, Wd_ih, bd, pred); }
  { int n = B_ * HIDc;   init_k<<<(n + TPB - 1) / TPB, TPB, 0, stream>>>(hbf0, hf32, cbuf); }

  for (int t = 0; t < T_; ++t) {
    unsigned short* hin  = (t & 1) ? hbf1 : hbf0;
    unsigned short* hout = (t & 1) ? hbf0 : hbf1;
    lstm_step_k<<<lgrid, TPB, 0, stream>>>(hin, wdhh_bf, cbuf, pred,
                                           nullptr, nullptr, nullptr, t, hout, hf32);
    yproj_k<<<B_ / 8, TPB, 0, stream>>>(hf32, Wpd, bpd, o_X, t);
  }
}
